// SelfAttention_15745350107814
// MI455X (gfx1250) — matrix operand order repaired and checked by reading, not compiled, not run on hardware
//
#include <hip/hip_runtime.h>
#include <hip/hip_bf16.h>
#include <stdint.h>

// Self-attention: x(8,32,32,512) -> QKV proj -> 8-head attention (N=1024, D=64) -> out proj.
// v_wmma_f32_16x16x32_f16 everywhere (f16 in, f32 accumulate), flash softmax,
// TDM tensor_load_to_lds for K tiles, global_load_async_to_lds_b128 for GEMM2 A tiles,
// global_prefetch for the next K-slab.

typedef __attribute__((ext_vector_type(16))) _Float16 v16h;
typedef __attribute__((ext_vector_type(8)))  _Float16 v8h;
typedef __attribute__((ext_vector_type(8)))  float    v8f;
typedef __attribute__((ext_vector_type(4)))  unsigned u32x4;
typedef __attribute__((ext_vector_type(8)))  int      i32x8;
typedef __attribute__((ext_vector_type(4)))  int      i32x4;

__device__ __forceinline__ v8f wmma_f16(v16h a, v16h b, v8f c) {
  // (neg_a, A, neg_b, B, c_mod, C, reuse_a, reuse_b)
  return __builtin_amdgcn_wmma_f32_16x16x32_f16(false, a, false, b, (short)0, c, false, false);
}

__device__ __forceinline__ v16h mk16(v8h a, v8h b) {
  return __builtin_shufflevector(a, b, 0,1,2,3,4,5,6,7,8,9,10,11,12,13,14,15);
}

// Low 32 bits of a flat LDS pointer == LDS byte offset (aperture truncation).
__device__ __forceinline__ unsigned lds_off(const void* p) {
  return (unsigned)(uintptr_t)p;
}

// 16B global -> LDS via the async engine (ASYNCcnt), GV addressing mode.
__device__ __forceinline__ void async_cp16(const void* gsrc, unsigned ldst) {
  unsigned long long ga = (unsigned long long)(uintptr_t)gsrc;
  asm volatile("global_load_async_to_lds_b128 %0, %1, off"
               :: "v"(ldst), "v"(ga) : "memory");
}

// ---------------------------------------------------------------------------
// Tiled GEMM: C[M,N] = A[M,K] * W[K,N] + bias.  W fp32 (converted while staging
// to LDS).  A fp32 (A_F16=false) or f16 (A_F16=true, staged via async-LDS).
// Block = 128 thr = 4 waves, block tile 64x64, wave tile 32x32 (2x2 WMMA accs).
// ---------------------------------------------------------------------------
template <bool A_F16, bool OUT_F32>
__global__ void __launch_bounds__(128)
gemm_wmma(const float* __restrict__ Af, const _Float16* __restrict__ Ah,
          const float* __restrict__ W, const float* __restrict__ bias,
          _Float16* __restrict__ outH, float* __restrict__ outF,
          int M, int N, int K)
{
  constexpr int LDA = 48;   // 32 k + 16 pad (f16): 16B-aligned rows, fewer bank conflicts
  constexpr int LDB = 48;   // Bs stored transposed: [n][k]
  __shared__ __attribute__((aligned(32))) _Float16 As[64 * LDA];
  __shared__ __attribute__((aligned(32))) _Float16 Bs[64 * LDB];

  const int n0   = blockIdx.x * 64;
  const int m0   = blockIdx.y * 64;
  const int lane = threadIdx.x & 31;
  const int l15  = lane & 15;
  const int hi   = lane >> 4;
  const int wave = threadIdx.x >> 5;
  const int wm   = (wave >> 1) * 32;
  const int wn   = (wave & 1) * 32;

  v8f acc[4] = {};

  for (int k0 = 0; k0 < K; k0 += 32) {
    // --- stage A tile (64x32) ---
    {
      const int r   = threadIdx.x >> 1;        // 0..63
      const int seg = (threadIdx.x & 1) * 16;  // 0 or 16
      _Float16* dst = &As[r * LDA + seg];
      if constexpr (A_F16) {
        const _Float16* s = Ah + (size_t)(m0 + r) * K + k0 + seg;
        async_cp16(s,     lds_off(dst));       // ASYNCcnt path: no VGPR round-trip
        async_cp16(s + 8, lds_off(dst + 8));
      } else {
        const float* s = Af + (size_t)(m0 + r) * K + k0 + seg;
        #pragma unroll
        for (int i = 0; i < 16; ++i) dst[i] = (_Float16)s[i];
      }
    }
    // --- stage W tile transposed -> Bs[n][k] (32x64 in, 64x32 out) ---
    {
      const int kk   = threadIdx.x >> 2;        // 0..31
      const int nseg = (threadIdx.x & 3) * 16;  // 0,16,32,48
      const float* s = W + (size_t)(k0 + kk) * N + n0 + nseg;
      #pragma unroll
      for (int i = 0; i < 16; ++i) Bs[(nseg + i) * LDB + kk] = (_Float16)s[i];
    }
    // prefetch next K-slab of A (global_prefetch_b8)
    if (k0 + 32 < K) {
      const int r = threadIdx.x >> 1;
      if constexpr (A_F16) __builtin_prefetch(Ah + (size_t)(m0 + r) * K + k0 + 32, 0, 1);
      else                 __builtin_prefetch(Af + (size_t)(m0 + r) * K + k0 + 32, 0, 1);
    }
    if constexpr (A_F16) asm volatile("s_wait_asynccnt 0" ::: "memory");
    __syncthreads();

    // A frag: lane = row M=l15; elems 0-7 -> K = 8*hi+0..7, 8-15 -> K = 16+8*hi+0..7
    const _Float16* a0p = &As[(wm      + l15) * LDA + hi * 8];
    const _Float16* a1p = &As[(wm + 16 + l15) * LDA + hi * 8];
    v16h a0 = mk16(*(const v8h*)a0p, *(const v8h*)(a0p + 16));
    v16h a1 = mk16(*(const v8h*)a1p, *(const v8h*)(a1p + 16));
    // B frag: lane = col N=l15; elems 0-7 -> K = 8*hi+0..7, 8-15 -> K = 16+8*hi+0..7 in Bs[n][k]
    const _Float16* b0p = &Bs[(wn      + l15) * LDB + hi * 8];
    const _Float16* b1p = &Bs[(wn + 16 + l15) * LDB + hi * 8];
    v16h b0 = mk16(*(const v8h*)b0p, *(const v8h*)(b0p + 16));
    v16h b1 = mk16(*(const v8h*)b1p, *(const v8h*)(b1p + 16));

    acc[0] = wmma_f16(a0, b0, acc[0]);
    acc[1] = wmma_f16(a0, b1, acc[1]);
    acc[2] = wmma_f16(a1, b0, acc[2]);
    acc[3] = wmma_f16(a1, b1, acc[3]);

    __syncthreads();
  }

  // C/D layout: VGPR r holds (M = r + 8*hi, N = l15) within each 16x16 tile
  #pragma unroll
  for (int t = 0; t < 4; ++t) {
    const int mi  = t >> 1, ni = t & 1;
    const int gm0 = m0 + wm + mi * 16 + 8 * hi;
    const int gn  = n0 + wn + ni * 16 + l15;
    const float bv = bias[gn];
    #pragma unroll
    for (int r = 0; r < 8; ++r) {
      const float v = acc[t][r] + bv;
      if constexpr (OUT_F32) __builtin_nontemporal_store(v, &outF[(size_t)(gm0 + r) * N + gn]);
      else                   outH[(size_t)(gm0 + r) * N + gn] = (_Float16)v;
    }
  }
}

// ---------------------------------------------------------------------------
// Flash attention.  P = f16 QKV buffer [8192][1536] (q|k|v, head h at +h*64).
// Block = 8 waves (256 thr) handles 128 query rows of one (b,h).  K tile is
// DMA'd by the Tensor Data Mover (wave 0 builds D#, TDM pads rows 128B+16B to
// match the LDS stride of 72 f16); V tile staged transposed by all threads.
// S = Q*K^T (4 wmma), online f32 softmax (shfl_xor row reductions), P restaged
// via per-wave LDS (C->A layout, s_wait_dscnt), O += P*V (4 wmma) + rescale.
// ---------------------------------------------------------------------------
__global__ void __launch_bounds__(256)
attn_flash(const _Float16* __restrict__ P, _Float16* __restrict__ O)
{
  const int bh   = blockIdx.y;        // 0..63
  const int b    = bh >> 3;
  const int h    = bh & 7;
  const int qblk = blockIdx.x;        // 0..7  (128 rows each)
  const int wave = threadIdx.x >> 5;
  const int lane = threadIdx.x & 31;
  const int l15  = lane & 15;
  const int hi   = lane >> 4;

  __shared__ __attribute__((aligned(32))) _Float16 Ks [32 * 72];       // K tile [j][d], TDM-padded
  __shared__ __attribute__((aligned(32))) _Float16 Vts[64 * 40];       // V tile transposed [d][j]
  __shared__ __attribute__((aligned(32))) _Float16 Pst[8 * 16 * 40];   // per-wave P staging

  const _Float16* Pb = P + (size_t)b * 1024 * 1536 + (size_t)h * 64;
  const int qrow = qblk * 128 + wave * 16;     // first query row of this wave

  // Q fragments: qa[ds] covers d = ds*32 .. +31 (A layout, row M = l15)
  v16h qa[2];
  #pragma unroll
  for (int ds = 0; ds < 2; ++ds) {
    const _Float16* qp = Pb + (size_t)(qrow + l15) * 1536 + ds * 32 + hi * 8;
    qa[ds] = mk16(*(const v8h*)qp, *(const v8h*)(qp + 16));
  }

  float rmax[8], rsum[8];
  #pragma unroll
  for (int r = 0; r < 8; ++r) { rmax[r] = -1e30f; rsum[r] = 0.0f; }
  v8f oacc[4] = {};

  _Float16* Pw = &Pst[wave * 16 * 40];
  const unsigned ks_lds = lds_off(&Ks[0]);

  for (int jt = 0; jt < 32; ++jt) {
    // --- K tile via Tensor Data Mover: 2D tile 64(d) x 32(j), stride 1536 elems,
    //     pad 4 DWORDs after every 32 DWORDs stored -> LDS row stride 72 f16.
    if (wave == 0) {
      const uint64_t ga = (uint64_t)(uintptr_t)(Pb + (size_t)(jt * 32) * 1536 + 512);
      u32x4 g0;
      g0[0] = 1u;                                              // count=1, user desc
      g0[1] = ks_lds;                                          // lds_addr
      g0[2] = (unsigned)(ga & 0xFFFFFFFFu);                    // global_addr[31:0]
      g0[3] = (unsigned)((ga >> 32) & 0x01FFFFFFu) | 0x80000000u; // addr[56:32] | type=2
      i32x8 g1;
      g1[0] = (int)((1u << 16)        // data_size = 2 bytes
                  | (1u << 20)        // pad_enable
                  | (4u << 22)        // pad_interval: 32 DWORDs (=128B row)
                  | (3u << 25));      // pad_amount: 4 DWORDs (=16B)
      g1[1] = (int)(64u << 16);       // tensor_dim0 = 64 (elems per head row)
      g1[2] = (int)(32u << 16);       // tensor_dim0 hi=0 | tensor_dim1 = 32 rows
      g1[3] = (int)(64u << 16);       // tensor_dim1 hi=0 | tile_dim0 = 64
      g1[4] = 32;                     // tile_dim1 = 32, tile_dim2 = 0
      g1[5] = 1536;                   // tensor_dim0_stride (elems)
      g1[6] = 0;
      g1[7] = 0;
      i32x4 gz4 = {0, 0, 0, 0};
      i32x8 gz8 = {0, 0, 0, 0, 0, 0, 0, 0};
      __builtin_amdgcn_tensor_load_to_lds(g0, g1, gz4, gz4, gz8, 0);
    }
    // --- V tile staged transposed by all 256 threads ---
    {
      const int j   = threadIdx.x >> 3;        // 0..31
      const int seg = (threadIdx.x & 7) * 8;   // 0..56
      const size_t row = (size_t)(jt * 32 + j) * 1536;
      v8h vv = *(const v8h*)(Pb + row + 1024 + seg);
      #pragma unroll
      for (int i = 0; i < 8; ++i) Vts[(seg + i) * 40 + j] = vv[i];
    }
    if (wave == 0) __builtin_amdgcn_s_wait_tensorcnt(0);
    __syncthreads();

    // S tiles: s0 = key cols 0..15, s1 = 16..31
    v8f s0 = {}, s1 = {};
    #pragma unroll
    for (int ds = 0; ds < 2; ++ds) {
      const _Float16* k0p = &Ks[ l15       * 72 + ds * 32 + hi * 8];
      const _Float16* k1p = &Ks[(16 + l15) * 72 + ds * 32 + hi * 8];
      v16h kb0 = mk16(*(const v8h*)k0p, *(const v8h*)(k0p + 16));
      v16h kb1 = mk16(*(const v8h*)k1p, *(const v8h*)(k1p + 16));
      s0 = wmma_f16(qa[ds], kb0, s0);
      s1 = wmma_f16(qa[ds], kb1, s1);
    }

    // online softmax; row m = r + 8*hi lives in VGPR r across 16 lanes (cols = l15)
    #pragma unroll
    for (int r = 0; r < 8; ++r) {
      float t0 = s0[r] * 0.125f;               // 1/sqrt(64)
      float t1 = s1[r] * 0.125f;
      float lm = fmaxf(t0, t1);
      lm = fmaxf(lm, __shfl_xor(lm, 1, 32));
      lm = fmaxf(lm, __shfl_xor(lm, 2, 32));
      lm = fmaxf(lm, __shfl_xor(lm, 4, 32));
      lm = fmaxf(lm, __shfl_xor(lm, 8, 32));
      const float mnew  = fmaxf(rmax[r], lm);
      const float alpha = __expf(rmax[r] - mnew);
      rmax[r] = mnew;
      const float p0 = __expf(t0 - mnew);
      const float p1 = __expf(t1 - mnew);
      float ps = p0 + p1;
      ps += __shfl_xor(ps, 1, 32);
      ps += __shfl_xor(ps, 2, 32);
      ps += __shfl_xor(ps, 4, 32);
      ps += __shfl_xor(ps, 8, 32);
      rsum[r] = rsum[r] * alpha + ps;
      #pragma unroll
      for (int t = 0; t < 4; ++t) oacc[t][r] *= alpha;
      const int m = r + 8 * hi;
      Pw[m * 40 +      l15] = (_Float16)p0;
      Pw[m * 40 + 16 + l15] = (_Float16)p1;
    }
    // same-wave LDS RAW: drain ds_stores before the A-layout gather
    asm volatile("s_wait_dscnt 0" ::: "memory");

    // P as A fragment (16 rows x 32 j), V^T as B fragments per d-group
    const _Float16* pp = &Pw[l15 * 40 + hi * 8];
    v16h pa = mk16(*(const v8h*)pp, *(const v8h*)(pp + 16));
    #pragma unroll
    for (int t = 0; t < 4; ++t) {
      const _Float16* vq = &Vts[(t * 16 + l15) * 40 + hi * 8];
      v16h vb = mk16(*(const v8h*)vq, *(const v8h*)(vq + 16));
      oacc[t] = wmma_f16(pa, vb, oacc[t]);
    }
    __syncthreads();   // before next iteration's TDM/stores overwrite Ks/Vts
  }

  // normalize and write O (f16) as [b*1024+n][512] with head offset
  #pragma unroll
  for (int t = 0; t < 4; ++t) {
    #pragma unroll
    for (int r = 0; r < 8; ++r) {
      const int m = r + 8 * hi;
      const float v = oacc[t][r] / rsum[r];
      O[((size_t)b * 1024 + qrow + m) * 512 + h * 64 + t * 16 + l15] = (_Float16)v;
    }
  }
}

extern "C" void kernel_launch(void* const* d_in, const int* in_sizes, int n_in,
                              void* d_out, int out_size, void* d_ws, size_t ws_size,
                              hipStream_t stream) {
  (void)in_sizes; (void)n_in; (void)out_size; (void)ws_size;
  const float* x  = (const float*)d_in[0];   // 8192 x 512
  const float* Wi = (const float*)d_in[1];   // 512 x 1536
  const float* bi = (const float*)d_in[2];   // 1536
  const float* Wo = (const float*)d_in[3];   // 512 x 512
  const float* bo = (const float*)d_in[4];   // 512
  float* out = (float*)d_out;                // 8192 x 512 fp32

  _Float16* Ph = (_Float16*)d_ws;                  // QKV buffer: 8192*1536 f16 (24 MB)
  _Float16* Oh = Ph + (size_t)8192 * 1536;         // O buffer:   8192*512  f16 ( 8 MB)

  // 1) QKV projection: Ph = f16(x @ W_in + b_in)
  gemm_wmma<false, false><<<dim3(1536 / 64, 8192 / 64), dim3(128), 0, stream>>>(
      x, nullptr, Wi, bi, Ph, nullptr, 8192, 1536, 512);

  // 2) flash attention per (b,h), 128 query rows per block
  attn_flash<<<dim3(8, 64), dim3(256), 0, stream>>>(Ph, Oh);

  // 3) output projection: out = Oh @ W_out + b_out (fp32, non-temporal stores)
  gemm_wmma<true, true><<<dim3(512 / 64, 8192 / 64), dim3(128), 0, stream>>>(
      nullptr, Oh, Wo, bo, nullptr, out, 8192, 512, 512);
}
